// FP8PerChannelLinear_42649025249876
// MI455X (gfx1250) — matrix operand order repaired and checked by reading, not compiled, not run on hardware
//
#include <hip/hip_runtime.h>
#include <hip/hip_bf16.h>
#include <math.h>

typedef __attribute__((ext_vector_type(16))) _Float16 v16h;
typedef __attribute__((ext_vector_type(8)))  _Float16 v8h;
typedef __attribute__((ext_vector_type(4)))  _Float16 v4h;
typedef __attribute__((ext_vector_type(8)))  float    v8f;

#define FP8_MAXV 448.0f
#define EPSV     1e-12f

#define BM 128
#define BN 128
#define BK 32
#define LDA 48   // padded LDS row stride in halves (96 bytes, 16B aligned)
#define LDB 48

__device__ __forceinline__ float warp_reduce_max(float v) {
#pragma unroll
  for (int off = 16; off > 0; off >>= 1)
    v = fmaxf(v, __shfl_xor(v, off, 32));
  return v;
}

__device__ __forceinline__ float quantize1(float v, float inv_s) {
  return rintf(fminf(fmaxf(v * inv_s, -FP8_MAXV), FP8_MAXV));
}

// CDNA5 async global->LDS copy (ASYNCcnt-tracked), GV addressing mode.
// %0 = VGPR with LDS byte address, %1 = VGPR pair with 64-bit global address.
__device__ __forceinline__ void async_copy_b128(void* lds_ptr, const void* gptr) {
  unsigned lds_addr = (unsigned)(size_t)lds_ptr;           // LDS aperture low 32b = LDS offset
  unsigned long long gaddr = (unsigned long long)(size_t)gptr;
  asm volatile("global_load_async_to_lds_b128 %0, %1, off"
               :: "v"(lds_addr), "v"(gaddr) : "memory");
}

__device__ __forceinline__ void wait_asynccnt0() {
  asm volatile("s_wait_asynccnt 0x0" ::: "memory");
}

// ---------------- Kernel 1: global amax of x ----------------
__global__ __launch_bounds__(256)
void fp8lin_amax_kernel(const float* __restrict__ x, long long n4,
                        unsigned int* __restrict__ amax_bits) {
  __shared__ float smax[8];
  float m = 0.0f;
  long long i = (long long)blockIdx.x * blockDim.x + threadIdx.x;
  long long stride = (long long)gridDim.x * blockDim.x;
  const float4* x4 = (const float4*)x;
  for (; i < n4; i += stride) {
    float4 v = x4[i];
    m = fmaxf(m, fmaxf(fmaxf(fabsf(v.x), fabsf(v.y)),
                       fmaxf(fabsf(v.z), fabsf(v.w))));
  }
  m = warp_reduce_max(m);
  int wid = threadIdx.x >> 5, lane = threadIdx.x & 31;
  if (lane == 0) smax[wid] = m;
  __syncthreads();
  if (wid == 0) {
    m = (lane < 8) ? smax[lane] : 0.0f;
    m = warp_reduce_max(m);
    if (lane == 0) atomicMax(amax_bits, __float_as_uint(m)); // x>=0: bit order == float order
  }
}

// ---------------- Kernel 2: per-row weight quant -> fp16 ----------------
__global__ __launch_bounds__(256)
void fp8lin_wquant_kernel(const float* __restrict__ w, int K,
                          const unsigned int* __restrict__ amax_bits,
                          _Float16* __restrict__ wq,
                          float* __restrict__ combo) {
  __shared__ float smax[8];
  __shared__ float s_inv;
  int o = blockIdx.x;
  const float* wr = w + (size_t)o * K;
  float m = 0.0f;
  for (int k = threadIdx.x; k < K; k += blockDim.x) m = fmaxf(m, fabsf(wr[k]));
  m = warp_reduce_max(m);
  int wid = threadIdx.x >> 5, lane = threadIdx.x & 31;
  if (lane == 0) smax[wid] = m;
  __syncthreads();
  if (threadIdx.x == 0) {
    float wm = 0.0f;
#pragma unroll
    for (int i = 0; i < 8; ++i) wm = fmaxf(wm, smax[i]);
    float wscale = fmaxf(wm / FP8_MAXV, EPSV);
    float in_scale = fmaxf(__uint_as_float(*amax_bits) / FP8_MAXV, EPSV);
    combo[o] = in_scale * wscale;       // fused dequant scale for this out-channel
    s_inv = 1.0f / wscale;
  }
  __syncthreads();
  float inv = s_inv;
  _Float16* wqr = wq + (size_t)o * K;
  for (int k = threadIdx.x; k < K; k += blockDim.x)
    wqr[k] = (_Float16)quantize1(wr[k], inv);
}

// ---------------- Kernel 3: double-buffered WMMA GEMM ----------------
// A (x) quantized fp32->fp16 on the fly; B (wq) streamed global->LDS via the
// CDNA5 async-to-LDS path (ASYNCcnt), overlapped with WMMA on the other buffer.
__global__ __launch_bounds__(256)
void fp8lin_gemm_kernel(const float* __restrict__ x,
                        const _Float16* __restrict__ wq,
                        const float* __restrict__ combo,
                        const float* __restrict__ bias,
                        const unsigned int* __restrict__ amax_bits,
                        float* __restrict__ out,
                        int M, int N, int K) {
  __shared__ _Float16 ldsA[2][BM * LDA];
  __shared__ _Float16 ldsB[2][BN * LDB];

  const int tid    = threadIdx.x;
  const int lane   = tid & 31;
  const int waveId = tid >> 5;
  const int waveM  = waveId & 3;   // 4 waves along M, 32 rows each
  const int waveN  = waveId >> 2;  // 2 waves along N, 64 cols each

  const int rowBase = blockIdx.y * BM;
  const int colBase = blockIdx.x * BN;

  const float in_scale = fmaxf(__uint_as_float(*amax_bits) / FP8_MAXV, EPSV);
  const float inv_s = 1.0f / in_scale;

  v8f acc[2][4] = {};
  float4 aReg[4];

  auto loadA_global = [&](int k0) {
#pragma unroll
    for (int i = 0; i < 4; ++i) {
      int idx = tid + i * 256;       // 0..1023 float4 slots (8 per row)
      int r   = idx >> 3;
      int c4  = idx & 7;
      aReg[i] = *(const float4*)(x + (size_t)(rowBase + r) * K + k0 + c4 * 4);
    }
  };
  auto storeA_lds = [&](int buf) {
#pragma unroll
    for (int i = 0; i < 4; ++i) {
      int idx = tid + i * 256;
      int r   = idx >> 3;
      int c4  = idx & 7;
      v4h h;
      h.x = (_Float16)quantize1(aReg[i].x, inv_s);
      h.y = (_Float16)quantize1(aReg[i].y, inv_s);
      h.z = (_Float16)quantize1(aReg[i].z, inv_s);
      h.w = (_Float16)quantize1(aReg[i].w, inv_s);
      *(v4h*)&ldsA[buf][r * LDA + c4 * 4] = h;
    }
  };
  auto loadB_tile = [&](int buf, int k0) {
#pragma unroll
    for (int i = 0; i < 2; ++i) {
      int idx = tid + i * 256;       // 0..511 chunks of 8 halves (4 per row)
      int r   = idx >> 2;
      int c8  = idx & 3;
      const _Float16* g = wq + (size_t)(colBase + r) * K + k0 + c8 * 8;
      _Float16* l = &ldsB[buf][r * LDB + c8 * 8];
      async_copy_b128(l, g);
    }
  };

  const int nk = K / BK;

  // ---- prologue: fill buffer 0 ----
  loadB_tile(0, 0);
  loadA_global(0);
  storeA_lds(0);
  wait_asynccnt0();
  __syncthreads();

  for (int kt = 0; kt < nk; ++kt) {
    const int cur = kt & 1;
    const int nxt = cur ^ 1;
    const bool more = (kt + 1 < nk);   // block-uniform

    if (more) {
      loadB_tile(nxt, (kt + 1) * BK);  // async: overlaps with WMMA below
      loadA_global((kt + 1) * BK);     // global loads in flight during compute
    }

    // ---- fragment gathers per documented 16-bit WMMA layouts ----
    v16h aF[2], bF[4];
#pragma unroll
    for (int mt = 0; mt < 2; ++mt) {
      const _Float16* ap =
          &ldsA[cur][(waveM * 32 + mt * 16 + (lane & 15)) * LDA + ((lane >> 4) << 3)];
      v8h lo = *(const v8h*)ap;        // K = kb .. kb+7
      v8h hi = *(const v8h*)(ap + 16); // K = kb+16 .. kb+23
      aF[mt] = __builtin_shufflevector(lo, hi, 0,1,2,3,4,5,6,7,8,9,10,11,12,13,14,15);
    }
#pragma unroll
    for (int nt = 0; nt < 4; ++nt) {
      const _Float16* bp =
          &ldsB[cur][(waveN * 64 + nt * 16 + (lane & 15)) * LDB + ((lane >> 4) << 3)];
      v8h lo = *(const v8h*)bp;        // K = kb .. kb+7
      v8h hi = *(const v8h*)(bp + 16); // K = kb+16 .. kb+23
      bF[nt] = __builtin_shufflevector(lo, hi, 0,1,2,3,4,5,6,7,8,9,10,11,12,13,14,15);
    }

#pragma unroll
    for (int mt = 0; mt < 2; ++mt)
#pragma unroll
      for (int nt = 0; nt < 4; ++nt)
        acc[mt][nt] = __builtin_amdgcn_wmma_f32_16x16x32_f16(
            false, aF[mt], false, bF[nt], (short)0, acc[mt][nt], false, false);

    if (more) storeA_lds(nxt);         // quantize+stage next A while WMMAs drain
    wait_asynccnt0();                  // own async writes complete before signaling
    __syncthreads();                   // one barrier per K-step (double-buffered)
  }

  // ---- epilogue: dequant + bias; C/D layout: VGPR r -> M=r (+8 for lanes>=16), lane%16 -> N ----
#pragma unroll
  for (int mt = 0; mt < 2; ++mt) {
    int r0 = rowBase + waveM * 32 + mt * 16 + ((lane >> 4) << 3);
#pragma unroll
    for (int nt = 0; nt < 4; ++nt) {
      int c = colBase + waveN * 64 + nt * 16 + (lane & 15);
      float cs = combo[c];
      float bv = bias[c];
      float* op = out + (size_t)r0 * N + c;
#pragma unroll
      for (int r = 0; r < 8; ++r)
        op[(size_t)r * N] = acc[mt][nt][r] * cs + bv;
    }
  }
}

extern "C" void kernel_launch(void* const* d_in, const int* in_sizes, int n_in,
                              void* d_out, int out_size, void* d_ws, size_t ws_size,
                              hipStream_t stream) {
  const float* x    = (const float*)d_in[0];
  const float* w    = (const float*)d_in[1];
  const float* bias = (const float*)d_in[2];
  float* out = (float*)d_out;

  const int N = in_sizes[2];                 // 2048 out channels
  const int K = in_sizes[1] / N;             // 2048
  const long long MK = (long long)in_sizes[0];
  const int M = (int)(MK / K);               // 16384

  // workspace layout: [amax u32 | pad to 16 | combo f32[N] | wq f16[N*K]]
  unsigned int* amax_bits = (unsigned int*)d_ws;
  float* combo = (float*)((char*)d_ws + 16);
  size_t wq_off = 16 + (((size_t)N * sizeof(float) + 15) & ~(size_t)15);
  _Float16* wqp = (_Float16*)((char*)d_ws + wq_off);

  (void)hipMemsetAsync(amax_bits, 0, sizeof(unsigned int), stream);

  long long n4 = MK / 4;
  fp8lin_amax_kernel<<<2048, 256, 0, stream>>>(x, n4, amax_bits);
  fp8lin_wquant_kernel<<<N, 256, 0, stream>>>(w, K, amax_bits, wqp, combo);

  dim3 grid(N / BN, M / BM);  // (16, 128); N-tiles adjacent -> x rows reused via L2
  fp8lin_gemm_kernel<<<grid, 256, 0, stream>>>(x, wqp, combo, bias, amax_bits,
                                               out, M, N, K);
}
